// Block_12120397709595
// MI455X (gfx1250) — matrix operand order repaired and checked by reading, not compiled, not run on hardware
//
#include <hip/hip_runtime.h>
#include <math.h>

// ---------------- types ----------------
typedef __attribute__((ext_vector_type(16))) _Float16 v16h;
typedef __attribute__((ext_vector_type(8)))  float    v8f;
typedef _Float16 f16;

struct alignas(16) H8 { _Float16 h[8]; };

// Problem constants
#define BB   8
#define TT   1024
#define CC   1024
#define HH   16
#define DD   64
#define FFF  4096
#define MM   (BB*TT)      // 8192 rows

__device__ __forceinline__ v8f wmma16(v16h a, v16h b, v8f c) {
  return __builtin_amdgcn_wmma_f32_16x16x32_f16(false, a, false, b, (short)0, c, false, false);
}

// A-operand (16x32 f16): lane<16 -> K [0..15]; lane>=16 -> K [16..31]
__device__ __forceinline__ v16h loadA16(const f16* rowk, int koffl) {
  H8 a = *(const H8*)(rowk + 2 * koffl);
  H8 b = *(const H8*)(rowk + 2 * koffl + 8);
  v16h r;
#pragma unroll
  for (int i = 0; i < 8; i++) { r[i] = a.h[i]; r[i + 8] = b.h[i]; }
  return r;
}
// 16 contiguous halfs (B-operand per lane), 16B-aligned is enough
__device__ __forceinline__ v16h load16(const f16* p) {
  H8 a = *(const H8*)p;
  H8 b = *(const H8*)(p + 8);
  v16h r;
#pragma unroll
  for (int i = 0; i < 8; i++) { r[i] = a.h[i]; r[i + 8] = b.h[i]; }
  return r;
}

__device__ __forceinline__ float rmax16(float v) {
#pragma unroll
  for (int m = 1; m < 16; m <<= 1) v = fmaxf(v, __shfl_xor(v, m));
  return v;
}
__device__ __forceinline__ float rsum16(float v) {
#pragma unroll
  for (int m = 1; m < 16; m <<= 1) v += __shfl_xor(v, m);
  return v;
}

// ---------------- weight convert + transpose: W (K x N, f32) -> Wt (N x K, f16) ----------------
__global__ __launch_bounds__(256) void convT(const float* __restrict__ W, f16* __restrict__ Wt,
                                             int K, int N) {
  __shared__ float tile[32][33];
  int n0 = blockIdx.x * 32, k0 = blockIdx.y * 32;
  int tx = threadIdx.x & 31, ty = threadIdx.x >> 5; // ty: 0..7
#pragma unroll
  for (int i = 0; i < 32; i += 8)
    tile[ty + i][tx] = W[(size_t)(k0 + ty + i) * N + (n0 + tx)];
  __syncthreads();
#pragma unroll
  for (int i = 0; i < 32; i += 8)
    Wt[(size_t)(n0 + ty + i) * K + (k0 + tx)] = (f16)tile[tx][ty + i];
}

// ---------------- layernorm: f32 in -> f16 out ----------------
__global__ __launch_bounds__(256) void ln_f16(const float* __restrict__ x,
                                              const float* __restrict__ g,
                                              const float* __restrict__ b,
                                              f16* __restrict__ out, int C) {
  int row = blockIdx.x;
  const float* xr = x + (size_t)row * C;
  float s = 0.f, s2 = 0.f;
  for (int c = threadIdx.x; c < C; c += blockDim.x) { float v = xr[c]; s += v; s2 += v * v; }
  __shared__ float sh[2][8];
#pragma unroll
  for (int m = 1; m < 32; m <<= 1) { s += __shfl_xor(s, m); s2 += __shfl_xor(s2, m); }
  int wid = threadIdx.x >> 5, lane = threadIdx.x & 31;
  if (lane == 0) { sh[0][wid] = s; sh[1][wid] = s2; }
  __syncthreads();
  if (wid == 0) {
    s  = (lane < 8) ? sh[0][lane] : 0.f;
    s2 = (lane < 8) ? sh[1][lane] : 0.f;
#pragma unroll
    for (int m = 1; m < 8; m <<= 1) { s += __shfl_xor(s, m); s2 += __shfl_xor(s2, m); }
    if (lane == 0) { sh[0][0] = s; sh[1][0] = s2; }
  }
  __syncthreads();
  float mu   = sh[0][0] / C;
  float var  = sh[1][0] / C - mu * mu;
  float rstd = rsqrtf(var + 1e-5f);
  for (int c = threadIdx.x; c < C; c += blockDim.x)
    out[(size_t)row * C + c] = (f16)((xr[c] - mu) * rstd * g[c] + b[c]);
}

// ---------------- WMMA GEMM: out(MxN) = A(MxK,f16) * Wt(NxK,f16)^T + bias [+epilogue] ----------------
// wave tile: 32 (M) x 64 (N), K-step 32
enum { EPI_F16 = 0, EPI_RES_F32 = 1, EPI_GELU_F16 = 2 };

template <int EPI>
__global__ __launch_bounds__(256) void gemm_f16(const f16* __restrict__ A,
                                                const f16* __restrict__ Bt,
                                                const float* __restrict__ bias,
                                                const float* __restrict__ resid,
                                                f16* __restrict__ outh,
                                                float* __restrict__ outf,
                                                int M, int N, int K) {
  int lane = threadIdx.x & 31;
  int wid  = (int)((blockIdx.x * blockDim.x + threadIdx.x) >> 5);
  int tilesN = N >> 6;
  int tm = wid / tilesN, tn = wid % tilesN;
  if (tm * 32 >= M) return;
  int row0 = tm * 32, col0 = tn * 64;
  int l16   = lane & 15;
  int koffl = (lane < 16) ? 0 : 8;   // A chunk select
  int kbB   = (lane < 16) ? 0 : 16;  // B chunk select

  const f16* a0p = A + (size_t)(row0 + l16) * K;
  const f16* a1p = A + (size_t)(row0 + 16 + l16) * K;
  const f16* bp0 = Bt + (size_t)(col0 +  0 + l16) * K + kbB;
  const f16* bp1 = Bt + (size_t)(col0 + 16 + l16) * K + kbB;
  const f16* bp2 = Bt + (size_t)(col0 + 32 + l16) * K + kbB;
  const f16* bp3 = Bt + (size_t)(col0 + 48 + l16) * K + kbB;

  v8f acc[2][4];
#pragma unroll
  for (int i = 0; i < 2; i++)
#pragma unroll
    for (int j = 0; j < 4; j++)
#pragma unroll
      for (int r = 0; r < 8; r++) acc[i][j][r] = 0.f;

  for (int k = 0; k < K; k += 32) {
    v16h a0 = loadA16(a0p + k, koffl);
    v16h a1 = loadA16(a1p + k, koffl);
    v16h b0 = load16(bp0 + k);
    v16h b1 = load16(bp1 + k);
    v16h b2 = load16(bp2 + k);
    v16h b3 = load16(bp3 + k);
    acc[0][0] = wmma16(a0, b0, acc[0][0]);
    acc[1][0] = wmma16(a1, b0, acc[1][0]);
    acc[0][1] = wmma16(a0, b1, acc[0][1]);
    acc[1][1] = wmma16(a1, b1, acc[1][1]);
    acc[0][2] = wmma16(a0, b2, acc[0][2]);
    acc[1][2] = wmma16(a1, b2, acc[1][2]);
    acc[0][3] = wmma16(a0, b3, acc[0][3]);
    acc[1][3] = wmma16(a1, b3, acc[1][3]);
  }

  int rhalf = (lane < 16) ? 0 : 8;
#pragma unroll
  for (int mi = 0; mi < 2; mi++) {
#pragma unroll
    for (int ni = 0; ni < 4; ni++) {
      int col = col0 + ni * 16 + l16;
      float bv = bias[col];
#pragma unroll
      for (int r = 0; r < 8; r++) {
        int row = row0 + mi * 16 + r + rhalf;
        float v = acc[mi][ni][r] + bv;
        if (EPI == EPI_GELU_F16) v = 0.5f * v * (1.f + erff(v * 0.70710678118f));
        if (EPI == EPI_RES_F32) {
          outf[(size_t)row * N + col] = v + resid[(size_t)row * N + col];
        } else {
          outh[(size_t)row * N + col] = (f16)v;
        }
      }
    }
  }
}

// ---------------- causal attention (wei = k.q^T, softmax over q-index, out = wei @ v) ----------------
// grid: B*H*(T/128) blocks, 256 threads = 8 waves, wave w handles rows rb*128 + w*16 .. +15
#define VTS 136  // padded LDS stride for V^T rows (halfs), keeps 16B alignment
__global__ __launch_bounds__(256) void attn_kernel(const f16* __restrict__ kqv,
                                                   f16* __restrict__ out) {
  const int C3 = 3 * CC;
  int bid = blockIdx.x;
  int rb  = bid & 7;        // T/128 = 8 row blocks
  int bh  = bid >> 3;
  int h   = bh & (HH - 1);
  int b   = bh >> 4;
  int lane = threadIdx.x & 31;
  int w    = threadIdx.x >> 5;
  int l16  = lane & 15;
  int rhalf = (lane < 16) ? 0 : 8;
  int koffl = (lane < 16) ? 0 : 8;
  int kbB   = (lane < 16) ? 0 : 16;

  __shared__ f16 Vt[64 * VTS];       // V^T chunk: [d][j], 128 columns
  __shared__ f16 Ps[8][16 * 32];     // per-wave P scratch (16x32)

  int rowbase = rb * 128 + w * 16;
  int hcol = h * DD;
  const f16* krow = kqv + (size_t)(b * TT + rowbase + l16) * C3 + hcol;      // k split @ col 0
  v16h aK0 = loadA16(krow, koffl);        // d = 0..31
  v16h aK1 = loadA16(krow + 32, koffl);   // d = 32..63

  v8f O[4];
  v8f mrow, lacc;
#pragma unroll
  for (int r = 0; r < 8; r++) { mrow[r] = -INFINITY; lacc[r] = 0.f; }
#pragma unroll
  for (int dt = 0; dt < 4; dt++)
#pragma unroll
    for (int r = 0; r < 8; r++) O[dt][r] = 0.f;

  const float scale = 0.125f;  // 1/sqrt(64)

  for (int jc = 0; jc <= rb; ++jc) {
    int jstart = jc * 128;
    __syncthreads();
    // stage V^T chunk: Vt[d][j] = v[jstart+j][d]
    for (int idx = threadIdx.x; idx < 128 * 64; idx += 256) {
      int j = idx >> 6, d = idx & 63;
      Vt[d * VTS + j] = kqv[(size_t)(b * TT + jstart + j) * C3 + 2 * CC + hcol + d];
    }
    __syncthreads();

    int nsub = (jc == rb) ? (((w * 16 + 15) >> 5) + 1) : 4;
    for (int js = 0; js < nsub; ++js) {
      int jb = jstart + js * 32;
      // ---- S = K(16x64) x Q^T(64x32), two 16x16 column subtiles
      const f16* q0 = kqv + (size_t)(b * TT + jb + l16) * C3 + CC + hcol;        // q split @ col C
      const f16* q1 = kqv + (size_t)(b * TT + jb + 16 + l16) * C3 + CC + hcol;
      v8f s0, s1;
#pragma unroll
      for (int r = 0; r < 8; r++) { s0[r] = 0.f; s1[r] = 0.f; }
      s0 = wmma16(aK0, load16(q0 + kbB), s0);
      s0 = wmma16(aK1, load16(q0 + 32 + kbB), s0);
      s1 = wmma16(aK0, load16(q1 + kbB), s1);
      s1 = wmma16(aK1, load16(q1 + 32 + kbB), s1);

      // ---- scale, causal mask, online softmax
      int c0 = jb + l16, c1 = jb + 16 + l16;
      v8f alphav;
#pragma unroll
      for (int r = 0; r < 8; r++) {
        int row = rowbase + r + rhalf;
        float v0 = s0[r] * scale; if (c0 > row) v0 = -INFINITY;
        float v1 = s1[r] * scale; if (c1 > row) v1 = -INFINITY;
        float mx   = rmax16(fmaxf(v0, v1));
        float mnew = fmaxf(mrow[r], mx);
        float alpha = __expf(mrow[r] - mnew);
        mrow[r] = mnew;
        float p0 = __expf(v0 - mnew);
        float p1 = __expf(v1 - mnew);
        s0[r] = p0; s1[r] = p1;
        lacc[r] = lacc[r] * alpha + rsum16(p0 + p1);
        alphav[r] = alpha;
      }
#pragma unroll
      for (int dt = 0; dt < 4; dt++)
#pragma unroll
        for (int r = 0; r < 8; r++) O[dt][r] *= alphav[r];

      // ---- C-layout P -> LDS -> A-layout (wave-private scratch; DS ops are in-order per wave)
#pragma unroll
      for (int r = 0; r < 8; r++) {
        int prow = r + rhalf;
        Ps[w][prow * 32 + l16]      = (f16)s0[r];
        Ps[w][prow * 32 + 16 + l16] = (f16)s1[r];
      }
      asm volatile("" ::: "memory");
      v16h aP = loadA16(&Ps[w][l16 * 32], koffl);

      // ---- O(16x64) += P(16x32) x V(32x64)
#pragma unroll
      for (int dt = 0; dt < 4; dt++) {
        v16h bv = load16(&Vt[(dt * 16 + l16) * VTS + js * 32 + kbB]);
        O[dt] = wmma16(aP, bv, O[dt]);
      }
    }
  }

  // ---- normalize and write (B,T,C) f16, head columns h*64..h*64+63
  size_t obase = (size_t)(b * TT + rowbase) * CC + hcol;
#pragma unroll
  for (int r = 0; r < 8; r++) {
    float inv = 1.0f / lacc[r];
    int row = r + rhalf;
#pragma unroll
    for (int dt = 0; dt < 4; dt++)
      out[obase + (size_t)row * CC + dt * 16 + l16] = (f16)(O[dt][r] * inv);
  }
}

// ---------------- host launcher ----------------
static inline size_t align_up(size_t v, size_t a) { return (v + a - 1) & ~(a - 1); }

extern "C" void kernel_launch(void* const* d_in, const int* in_sizes, int n_in,
                              void* d_out, int out_size, void* d_ws, size_t ws_size,
                              hipStream_t stream) {
  (void)in_sizes; (void)n_in; (void)out_size; (void)ws_size;
  const float* x           = (const float*)d_in[0];
  const float* ln1_g       = (const float*)d_in[1];
  const float* ln1_b       = (const float*)d_in[2];
  const float* attn_w      = (const float*)d_in[3];   // (C, 3C)
  const float* attn_b      = (const float*)d_in[4];
  const float* attn_proj_w = (const float*)d_in[5];   // (C, C)
  const float* attn_proj_b = (const float*)d_in[6];
  const float* ln2_g       = (const float*)d_in[7];
  const float* ln2_b       = (const float*)d_in[8];
  const float* fc_w        = (const float*)d_in[9];   // (C, FF)
  const float* fc_b        = (const float*)d_in[10];
  const float* mlp_proj_w  = (const float*)d_in[11];  // (FF, C)
  const float* mlp_proj_b  = (const float*)d_in[12];

  char* ws = (char*)d_ws;
  size_t off = 0;
  auto take = [&](size_t bytes) { size_t o = off; off = align_up(off + bytes, 256); return (void*)(ws + o); };

  f16*   wqkv_t = (f16*)take((size_t)(3 * CC) * CC * 2);   // (3C x C)
  f16*   wprj_t = (f16*)take((size_t)CC * CC * 2);         // (C x C)
  f16*   wfc_t  = (f16*)take((size_t)FFF * CC * 2);        // (FF x C)
  f16*   wmlp_t = (f16*)take((size_t)CC * FFF * 2);        // (C x FF)
  f16*   h16    = (f16*)take((size_t)MM * CC * 2);         // LN output (reused for ln1 & ln2)
  f16*   kqv16  = (f16*)take((size_t)MM * 3 * CC * 2);
  f16*   a16    = (f16*)take((size_t)MM * CC * 2);         // attention output
  float* x1     = (float*)take((size_t)MM * CC * 4);       // residual after attention
  f16*   m16    = (f16*)take((size_t)MM * FFF * 2);        // gelu(fc) output

  // weights: convert + transpose to f16 (N x K)
  convT<<<dim3(3 * CC / 32, CC / 32), 256, 0, stream>>>(attn_w, wqkv_t, CC, 3 * CC);
  convT<<<dim3(CC / 32, CC / 32),     256, 0, stream>>>(attn_proj_w, wprj_t, CC, CC);
  convT<<<dim3(FFF / 32, CC / 32),    256, 0, stream>>>(fc_w, wfc_t, CC, FFF);
  convT<<<dim3(CC / 32, FFF / 32),    256, 0, stream>>>(mlp_proj_w, wmlp_t, FFF, CC);

  // h = LN1(x)
  ln_f16<<<MM, 256, 0, stream>>>(x, ln1_g, ln1_b, h16, CC);

  // kqv = h @ attn_w + attn_b  -> f16
  {
    int blocks = (MM / 32) * (3 * CC / 64) / 8;
    gemm_f16<EPI_F16><<<blocks, 256, 0, stream>>>(h16, wqkv_t, attn_b, nullptr,
                                                  kqv16, nullptr, MM, 3 * CC, CC);
  }

  // attention
  attn_kernel<<<BB * HH * (TT / 128), 256, 0, stream>>>(kqv16, a16);

  // x1 = x + attn @ attn_proj_w + attn_proj_b  -> f32
  {
    int blocks = (MM / 32) * (CC / 64) / 8;
    gemm_f16<EPI_RES_F32><<<blocks, 256, 0, stream>>>(a16, wprj_t, attn_proj_b, x,
                                                      nullptr, x1, MM, CC, CC);
  }

  // h2 = LN2(x1)
  ln_f16<<<MM, 256, 0, stream>>>(x1, ln2_g, ln2_b, h16, CC);

  // m = gelu(h2 @ fc_w + fc_b) -> f16
  {
    int blocks = (MM / 32) * (FFF / 64) / 8;
    gemm_f16<EPI_GELU_F16><<<blocks, 256, 0, stream>>>(h16, wfc_t, fc_b, nullptr,
                                                       m16, nullptr, MM, FFF, CC);
  }

  // out = x1 + m @ mlp_proj_w + mlp_proj_b -> f32
  {
    int blocks = (MM / 32) * (CC / 64) / 8;
    gemm_f16<EPI_RES_F32><<<blocks, 256, 0, stream>>>(m16, wmlp_t, mlp_proj_b, x1,
                                                      nullptr, (float*)d_out, MM, CC, FFF);
  }
}
